// StandardAttention_85160611545412
// MI455X (gfx1250) — matrix operand order repaired and checked by reading, not compiled, not run on hardware
//
#include <hip/hip_runtime.h>
#include <hip/hip_bf16.h>

#define DIMC  1024
#define SEQN  2048
#define BATCH 2
#define HEADS 16
#define HDIM  64
// SCALE = HEAD_DIM^-0.5 = 1/8 ; fold log2(e) so softmax uses exp2
#define QSCALE (0.125f * 1.4426950408889634f)

typedef __attribute__((ext_vector_type(16))) __bf16 v16bf;
typedef __attribute__((ext_vector_type(8)))  float  v8f;

union Frag {
    v16bf bf;
    uint4 q[2];
    unsigned w[8];
};

__device__ __forceinline__ unsigned short f2bf(float x) {
    union { __bf16 b; unsigned short s; } u;
    u.b = (__bf16)x;                                   // hardware v_cvt, RNE
    return u.s;
}
__device__ __forceinline__ unsigned pk2(float lo, float hi) {
    union { __bf16 b[2]; unsigned u; } v;
    v.b[0] = (__bf16)lo; v.b[1] = (__bf16)hi;
    return v.u;
}
__device__ __forceinline__ v8f vzero() {
    v8f z; for (int i = 0; i < 8; ++i) z[i] = 0.0f; return z;
}

// Flat shared-pointer low 32 bits == wave-relative LDS byte offset (ISA 10.2)
__device__ __forceinline__ unsigned ldsAddr(const void* p) {
    return (unsigned)(uintptr_t)p;
}
// CDNA5 async direct-to-LDS copy, 16B per lane, tracked by ASYNCcnt
__device__ __forceinline__ void asyncCopy16(unsigned ldsOff, const void* gaddr) {
    asm volatile("global_load_async_to_lds_b128 %0, %1, off"
                 :: "v"(ldsOff), "v"(gaddr) : "memory");
}
#define WAIT_ASYNC(N) asm volatile("s_wait_asynccnt " #N ::: "memory")

// A-operand (16x32 bf16): lane = M row, halves at K = kbase+16g+{0..15} (contiguous 32B)
__device__ __forceinline__ Frag ldA(const unsigned short* base, int row, int stride, int kbase, int g) {
    const uint4* p = reinterpret_cast<const uint4*>(base + (size_t)row * stride + kbase + 16 * g);
    Frag f; f.q[0] = p[0]; f.q[1] = p[1]; return f;
}
// B-operand (32x16 bf16): lane = N col, halves at K = kbase+16g+{0..15} (contiguous 32B)
__device__ __forceinline__ Frag ldB(const unsigned short* base, int row, int stride, int kbase, int g) {
    const uint4* p = reinterpret_cast<const uint4*>(base + (size_t)row * stride + kbase + 16 * g);
    Frag f; f.q[0] = p[0]; f.q[1] = p[1]; return f;
}

__device__ __forceinline__ v8f bwmma(Frag a, Frag b, v8f c) {
    return __builtin_amdgcn_wmma_f32_16x16x32_bf16(false, a.bf, false, b.bf,
                                                   (short)0, c, false, false);
}

// -------------------------------------------------------------------------
// Kernel 0: one-shot fp32 -> bf16 convert (8 elems/thread)
// -------------------------------------------------------------------------
__global__ __launch_bounds__(256) void cvt_kernel(
    const float* __restrict__ src, unsigned short* __restrict__ dst, int n8)
{
    int i = blockIdx.x * 256 + threadIdx.x;
    if (i < n8) {
        float4 a = reinterpret_cast<const float4*>(src)[2 * i];
        float4 b = reinterpret_cast<const float4*>(src)[2 * i + 1];
        uint4 o;
        o.x = pk2(a.x, a.y); o.y = pk2(a.z, a.w);
        o.z = pk2(b.x, b.y); o.w = pk2(b.z, b.w);
        reinterpret_cast<uint4*>(dst)[i] = o;
    }
}

// -------------------------------------------------------------------------
// Kernel 1: QKV = x @ w_qkv^T + b_qkv.  256 thr, 128(M) x 128(N) block tile,
// wave tile 64x32 (8 wmma / K-step).  Async ping-pong LDS staging with
// compile-time buffer indices (keeps WMMA accumulators pinned in place).
// blockIdx.x>>3 selects Q / K / V region (block-uniform epilogue).
// -------------------------------------------------------------------------
__global__ __launch_bounds__(256) void qkv_kernel(
    const unsigned short* __restrict__ xb, const unsigned short* __restrict__ wb,
    const float* __restrict__ bqkv,
    unsigned short* __restrict__ Qb, unsigned short* __restrict__ Kb,
    unsigned short* __restrict__ Vb)
{
    __shared__ alignas(16) unsigned short As[2][128 * 40];
    __shared__ alignas(16) unsigned short Bs[2][128 * 40];
    const int t = threadIdx.x, lane = t & 31, wave = t >> 5;
    const int g = lane >> 4, lr = lane & 15;
    const int wm = wave >> 2, wn = wave & 3;
    const int mBase = blockIdx.y * 128;
    const int nBase = blockIdx.x * 128;
    const int t3 = blockIdx.x >> 3;                    // 0:Q 1:K 2:V (uniform)

    v8f acc[4][2];
    for (int i = 0; i < 4; ++i) for (int j = 0; j < 2; ++j) acc[i][j] = vzero();

    auto issue = [&](int buf, int kb) {
        for (int jj = 0; jj < 2; ++jj) {               // A: 128x32 halves, 2/thread
            int idx = jj * 256 + t, r = idx >> 2, c = idx & 3;
            asyncCopy16(ldsAddr(&As[buf][r * 40 + c * 8]),
                        xb + (size_t)(mBase + r) * DIMC + kb + c * 8);
        }
        for (int jj = 0; jj < 2; ++jj) {               // B: 128x32 halves, 2/thread
            int idx = jj * 256 + t, r = idx >> 2, c = idx & 3;
            asyncCopy16(ldsAddr(&Bs[buf][r * 40 + c * 8]),
                        wb + (size_t)(nBase + r) * DIMC + kb + c * 8);
        }
    };
    auto compute = [&](int buf) {
        Frag b0 = ldB(&Bs[buf][0], wn * 32 + lr,      40, 0, g);
        Frag b1 = ldB(&Bs[buf][0], wn * 32 + 16 + lr, 40, 0, g);
        for (int mt = 0; mt < 4; ++mt) {
            Frag a = ldA(&As[buf][0], wm * 64 + mt * 16 + lr, 40, 0, g);
            acc[mt][0] = bwmma(a, b0, acc[mt][0]);
            acc[mt][1] = bwmma(a, b1, acc[mt][1]);
        }
    };

    issue(0, 0);
    for (int k = 0; k < DIMC / 32; k += 2) {           // explicit ping-pong
        issue(1, (k + 1) * 32);
        WAIT_ASYNC(0x4);                               // buffer 0 landed
        __syncthreads();
        compute(0);
        __syncthreads();
        if (k + 2 < DIMC / 32) { issue(0, (k + 2) * 32); WAIT_ASYNC(0x4); }
        else                   { WAIT_ASYNC(0x0); }
        __syncthreads();
        compute(1);
        __syncthreads();
    }

    const int creg = (nBase & 1023) + wn * 32;         // column base inside region
    auto epi = [&](auto storeFn) {
        for (int mt = 0; mt < 4; ++mt)
            for (int tj = 0; tj < 2; ++tj) {
                const int cc = creg + tj * 16 + lr;
                const float bias = bqkv[(t3 << 10) + cc];
                const int h = cc >> 6, d = cc & 63;
                const int mrow = mBase + wm * 64 + mt * 16 + g * 8;
                for (int i = 0; i < 8; ++i) {
                    const int m = mrow + i, bb = m >> 11, n = m & 2047;
                    storeFn(bb * HEADS + h, n, d, acc[mt][tj][i] + bias);
                }
            }
    };
    if (t3 == 0)
        epi([&](int bh, int n, int d, float v) {
            Qb[((size_t)(bh * SEQN + n)) * HDIM + d] = f2bf(v * QSCALE); });
    else if (t3 == 1)
        epi([&](int bh, int n, int d, float v) {
            Kb[((size_t)(bh * SEQN + n)) * HDIM + d] = f2bf(v); });
    else
        epi([&](int bh, int n, int d, float v) {
            Vb[((size_t)bh * HDIM + d) * SEQN + n] = f2bf(v); });
}

// -------------------------------------------------------------------------
// Kernel 2: flash attention (S^T = K*Q^T trick; per-lane softmax stats;
// P^T re-packs straight into B-operand layout). Async ping-pong K/V tiles.
// -------------------------------------------------------------------------
__device__ __forceinline__ Frag packPB(v8f t0, v8f t1, int g) {
    unsigned u0[4], u1[4];
    for (int i = 0; i < 4; ++i) {
        u0[i] = pk2(t0[2 * i], t0[2 * i + 1]);
        u1[i] = pk2(t1[2 * i], t1[2 * i + 1]);
    }
    Frag f;
    for (int i = 0; i < 4; ++i) {
        unsigned x0 = __shfl_xor(u0[i], 16);
        unsigned x1 = __shfl_xor(u1[i], 16);
        f.w[i]     = g ? x1    : u0[i];   // halves 0..7  : keys 16g+0..7
        f.w[4 + i] = g ? u1[i] : x0;      // halves 8..15 : keys 16g+8..15
    }
    return f;
}

__global__ __launch_bounds__(128) void attn_kernel(
    const unsigned short* __restrict__ Qb, const unsigned short* __restrict__ Kb,
    const unsigned short* __restrict__ Vb, unsigned short* __restrict__ Ab)
{
    __shared__ alignas(16) unsigned short Ks[2][64 * 72];   // [key][d]
    __shared__ alignas(16) unsigned short Vs[2][64 * 72];   // [d][key] (V^T)
    const int t = threadIdx.x, lane = t & 31, wave = t >> 5;
    const int g = lane >> 4, lr = lane & 15;
    const int bh = blockIdx.y;
    const int qbase = blockIdx.x * 64 + wave * 16;
    const unsigned short* Qp = Qb + (size_t)bh * SEQN * HDIM;
    const unsigned short* Kp = Kb + (size_t)bh * SEQN * HDIM;
    const unsigned short* Vp = Vb + (size_t)bh * HDIM * SEQN;

    Frag bq0 = ldB(Qp, qbase + lr, HDIM, 0,  g);       // Q^T held in registers
    Frag bq1 = ldB(Qp, qbase + lr, HDIM, 32, g);

    v8f o[4];
    for (int i = 0; i < 4; ++i) o[i] = vzero();
    float mrun = -1e30f, lsum = 0.0f;

    auto issue = [&](int buf, int kb0) {
        for (int jj = 0; jj < 4; ++jj) {               // 1024 x 16B total (512 K + 512 V), 8/thread
            int idx = jj * 128 + t, r = idx >> 3, c = idx & 7;
            asyncCopy16(ldsAddr(&Ks[buf][r * 72 + c * 8]),
                        Kp + (size_t)(kb0 + r) * HDIM + c * 8);
            asyncCopy16(ldsAddr(&Vs[buf][r * 72 + c * 8]),
                        Vp + (size_t)r * SEQN + kb0 + c * 8);
        }
    };
    auto compute = [&](int buf) {
        v8f s[4];                                      // S^T tiles [16 keys x 16 q]
        for (int mt = 0; mt < 4; ++mt) {
            Frag a0 = ldA(&Ks[buf][0], mt * 16 + lr, 72, 0,  g);
            Frag a1 = ldA(&Ks[buf][0], mt * 16 + lr, 72, 32, g);
            v8f z = vzero();
            z = bwmma(a0, bq0, z);
            s[mt] = bwmma(a1, bq1, z);
        }
        float mloc = -1e30f;
        for (int mt = 0; mt < 4; ++mt)
            for (int e = 0; e < 8; ++e) mloc = fmaxf(mloc, s[mt][e]);
        mloc = fmaxf(mloc, __shfl_xor(mloc, 16));
        const float mnew  = fmaxf(mrun, mloc);
        const float alpha = exp2f(mrun - mnew);
        float rs = 0.0f;
        for (int mt = 0; mt < 4; ++mt)
            for (int e = 0; e < 8; ++e) {
                float p = exp2f(s[mt][e] - mnew);
                s[mt][e] = p; rs += p;
            }
        rs += __shfl_xor(rs, 16);
        lsum = lsum * alpha + rs;
        mrun = mnew;
        for (int dt = 0; dt < 4; ++dt)
            for (int e = 0; e < 8; ++e) o[dt][e] *= alpha;

        Frag bp0 = packPB(s[0], s[1], g);
        Frag bp1 = packPB(s[2], s[3], g);
        for (int dt = 0; dt < 4; ++dt) {               // O^T += V^T * P^T
            Frag av0 = ldA(&Vs[buf][0], dt * 16 + lr, 72, 0,  g);
            o[dt] = bwmma(av0, bp0, o[dt]);
            Frag av1 = ldA(&Vs[buf][0], dt * 16 + lr, 72, 32, g);
            o[dt] = bwmma(av1, bp1, o[dt]);
        }
    };

    issue(0, 0);
    for (int j = 0; j < SEQN / 64; j += 2) {           // explicit ping-pong
        issue(1, (j + 1) * 64);
        WAIT_ASYNC(0x8);
        __syncthreads();
        compute(0);
        __syncthreads();
        if (j + 2 < SEQN / 64) { issue(0, (j + 2) * 64); WAIT_ASYNC(0x8); }
        else                   { WAIT_ASYNC(0x0); }
        __syncthreads();
        compute(1);
        __syncthreads();
    }

    const float inv = 1.0f / lsum;
    const int b = bh >> 4, h = bh & 15, q = qbase + lr;
    for (int dt = 0; dt < 4; ++dt)
        for (int i = 0; i < 8; ++i) {
            const int d = dt * 16 + g * 8 + i;
            Ab[((size_t)(b * SEQN + q)) * DIMC + h * HDIM + d] = f2bf(o[dt][i] * inv);
        }
}

// -------------------------------------------------------------------------
// Kernel 3: out = attn_out @ w_proj^T + b_proj  (both operands already bf16)
// -------------------------------------------------------------------------
__global__ __launch_bounds__(256) void proj_kernel(
    const unsigned short* __restrict__ Ab, const unsigned short* __restrict__ wb,
    const float* __restrict__ bp, float* __restrict__ out)
{
    __shared__ alignas(16) unsigned short As[2][128 * 40];
    __shared__ alignas(16) unsigned short Bs[2][128 * 40];
    const int t = threadIdx.x, lane = t & 31, wave = t >> 5;
    const int g = lane >> 4, lr = lane & 15;
    const int wm = wave >> 2, wn = wave & 3;
    const int mBase = blockIdx.y * 128;
    const int nBase = blockIdx.x * 128;

    v8f acc[4][2];
    for (int i = 0; i < 4; ++i) for (int j = 0; j < 2; ++j) acc[i][j] = vzero();

    auto issue = [&](int buf, int kb) {
        for (int jj = 0; jj < 2; ++jj) {
            int idx = jj * 256 + t, r = idx >> 2, c = idx & 3;
            asyncCopy16(ldsAddr(&As[buf][r * 40 + c * 8]),
                        Ab + (size_t)(mBase + r) * DIMC + kb + c * 8);
        }
        for (int jj = 0; jj < 2; ++jj) {
            int idx = jj * 256 + t, r = idx >> 2, c = idx & 3;
            asyncCopy16(ldsAddr(&Bs[buf][r * 40 + c * 8]),
                        wb + (size_t)(nBase + r) * DIMC + kb + c * 8);
        }
    };
    auto compute = [&](int buf) {
        Frag b0 = ldB(&Bs[buf][0], wn * 32 + lr,      40, 0, g);
        Frag b1 = ldB(&Bs[buf][0], wn * 32 + 16 + lr, 40, 0, g);
        for (int mt = 0; mt < 4; ++mt) {
            Frag a = ldA(&As[buf][0], wm * 64 + mt * 16 + lr, 40, 0, g);
            acc[mt][0] = bwmma(a, b0, acc[mt][0]);
            acc[mt][1] = bwmma(a, b1, acc[mt][1]);
        }
    };

    issue(0, 0);
    for (int k = 0; k < DIMC / 32; k += 2) {
        issue(1, (k + 1) * 32);
        WAIT_ASYNC(0x4);
        __syncthreads();
        compute(0);
        __syncthreads();
        if (k + 2 < DIMC / 32) { issue(0, (k + 2) * 32); WAIT_ASYNC(0x4); }
        else                   { WAIT_ASYNC(0x0); }
        __syncthreads();
        compute(1);
        __syncthreads();
    }

    for (int mt = 0; mt < 4; ++mt)
        for (int tj = 0; tj < 2; ++tj) {
            const int c = nBase + wn * 32 + tj * 16 + lr;
            const float bias = bp[c];
            const int mrow = mBase + wm * 64 + mt * 16 + g * 8;
            for (int i = 0; i < 8; ++i)
                out[(size_t)(mrow + i) * DIMC + c] = acc[mt][tj][i] + bias;
        }
}

extern "C" void kernel_launch(void* const* d_in, const int* in_sizes, int n_in,
                              void* d_out, int out_size, void* d_ws, size_t ws_size,
                              hipStream_t stream) {
    const float* x     = (const float*)d_in[0];
    const float* wqkv  = (const float*)d_in[1];
    const float* bqkv  = (const float*)d_in[2];
    const float* wproj = (const float*)d_in[3];
    const float* bproj = (const float*)d_in[4];
    float* out = (float*)d_out;

    // workspace layout (bf16 halves); x-region is reused for attention output
    unsigned short* xab    = (unsigned short*)d_ws;            // 4 Mi  (x, later Ab)
    unsigned short* wqkvb  = xab    + (size_t)4 * 1024 * 1024; // 3 Mi
    unsigned short* wprojb = wqkvb  + (size_t)3 * 1024 * 1024; // 1 Mi
    unsigned short* Qb     = wprojb + (size_t)1 * 1024 * 1024; // 4 Mi
    unsigned short* Kb     = Qb     + (size_t)4 * 1024 * 1024; // 4 Mi
    unsigned short* Vb     = Kb     + (size_t)4 * 1024 * 1024; // 4 Mi -> 40 MB total

    cvt_kernel<<<dim3(2048), 256, 0, stream>>>(x,     xab,    4 * 1024 * 1024 / 8);
    cvt_kernel<<<dim3(1536), 256, 0, stream>>>(wqkv,  wqkvb,  3 * 1024 * 1024 / 8);
    cvt_kernel<<<dim3(512),  256, 0, stream>>>(wproj, wprojb, 1 * 1024 * 1024 / 8);

    qkv_kernel<<<dim3(24, 32), 256, 0, stream>>>(xab, wqkvb, bqkv, Qb, Kb, Vb);
    attn_kernel<<<dim3(SEQN / 64, BATCH * HEADS), 128, 0, stream>>>(Qb, Kb, Vb, xab);
    proj_kernel<<<dim3(8, 32), 256, 0, stream>>>(xab, wprojb, bproj, out);
}
